// BilinearMatrixSAE_83562883711556
// MI455X (gfx1250) — matrix operand order repaired and checked by reading, not compiled, not run on hardware
//
#include <hip/hip_runtime.h>

// BilinearMatrixSAE for MI455X (gfx1250, wave32, WMMA).
//   encode : pre = x_flat @ M_enc^T   (M_enc rank-1 per feature, built in LDS, bf16 WMMA)
//   topk   : per-row top-32 + relu, in-place sparsify of coeffs
//   decode : recon = sum_j val_j * v_dec ⊗ w_dec  == [64x32]@[32x64] per batch (f16 WMMA, K=32 exact)
//   mse    : fused into decode (block reduce + atomicAdd)
// v2: fragment LDS reads expressed as contiguous uint4 (ds_load_b128) instead of
//     scattered b32 pairs -> halves ds count and drops per-fragment address VALU.

#define NFEAT 32768
#define BATCH 1024
#define DK 64
#define DV 64
#define DKV 4096
#define TOPK 32

typedef __attribute__((ext_vector_type(16))) __bf16    v16bf;
typedef __attribute__((ext_vector_type(16))) _Float16  v16h;
typedef __attribute__((ext_vector_type(8)))  float     v8f;

struct U8x4 { unsigned int u[8]; };
static_assert(sizeof(U8x4) == 32, "frag size");

static __device__ __forceinline__ unsigned short f32_to_bf16(float f) {
  unsigned int x = __builtin_bit_cast(unsigned int, f);
  unsigned int r = x + 0x7FFFu + ((x >> 16) & 1u);   // round-to-nearest-even
  return (unsigned short)(r >> 16);
}

static __device__ __forceinline__ v16bf frag_from2(uint4 q0, uint4 q1) {
  U8x4 u;
  u.u[0] = q0.x; u.u[1] = q0.y; u.u[2] = q0.z; u.u[3] = q0.w;
  u.u[4] = q1.x; u.u[5] = q1.y; u.u[6] = q1.z; u.u[7] = q1.w;
  return __builtin_bit_cast(v16bf, u);
}
static __device__ __forceinline__ v16h frag_from2h(uint4 q0, uint4 q1) {
  U8x4 u;
  u.u[0] = q0.x; u.u[1] = q0.y; u.u[2] = q0.z; u.u[3] = q0.w;
  u.u[4] = q1.x; u.u[5] = q1.y; u.u[6] = q1.z; u.u[7] = q1.w;
  return __builtin_bit_cast(v16h, u);
}

// ---------------------------------------------------------------- kernel 1
// x fp32 -> bf16 (8 MB, stays L2-resident for the encode sweep); zero mse.
__global__ void k_convert_x(const float* __restrict__ x,
                            unsigned short* __restrict__ xbf,
                            float* __restrict__ mse) {
  int i = blockIdx.x * 256 + threadIdx.x;
  if (i < BATCH * DKV) xbf[i] = f32_to_bf16(x[i]);
  if (i == 0) *mse = 0.0f;
}

// ---------------------------------------------------------------- kernel 2
// One block = 16 features. Build the 16x4096 bf16 slice of M_enc in LDS once,
// then each of the 8 waves sweeps 8 batch-tiles with v_wmma_f32_16x16x32_bf16.
__global__ void __launch_bounds__(256)
k_encode(const unsigned short* __restrict__ xbf,
         const float* __restrict__ Venc, const float* __restrict__ Wenc,
         const float* __restrict__ benc, float* __restrict__ pre) {
  extern __shared__ unsigned short smem[];          // [16][4096] bf16 = 128 KB
  const int f0  = blockIdx.x * 16;
  const int tid = threadIdx.x;

  // rank-1 strip: M_enc[f][k*64+v] = V_enc[f][k] * W_enc[f][v]
  for (int e = tid; e < 16 * DKV; e += 256) {
    int fi = e >> 12, kv = e & 4095, k = kv >> 6, v = kv & 63;
    float m = Venc[(f0 + fi) * DK + k] * Wenc[(f0 + fi) * DV + v];
    smem[fi * DKV + kv] = f32_to_bf16(m);
  }
  __syncthreads();

  const int wave = tid >> 5, lane = tid & 31, ln = lane & 15, lh = lane >> 4;
  v8f acc[8] = {};                                   // 8 batch-tiles per wave

  // B column ln: lane's 8 B-frag dwords are contiguous at uint index
  // (kc*16 + lh*16)/2, 16B aligned -> two ds_load_b128 per k-chunk.
  const uint4* brow = (const uint4*)(smem + ln * DKV + lh * 16);

  for (int kc = 0; kc < DKV / 32; ++kc) {
    const int kbase = kc * 32;
    uint4 bq0 = brow[kc * 2 + 0];
    uint4 bq1 = brow[kc * 2 + 1];
    v16bf bfrag = frag_from2(bq0, bq1);

#pragma unroll
    for (int s = 0; s < 8; ++s) {
      // A fragment (16x32 bf16): lane row = btile*16+ln, two global b128 loads
      int br = (wave * 8 + s) * 16 + ln;
      const uint4* ap = (const uint4*)(xbf + (size_t)br * DKV + kbase + lh * 16);
      v16bf afrag = frag_from2(ap[0], ap[1]);        // K 0..15 (+half*16)
      acc[s] = __builtin_amdgcn_wmma_f32_16x16x32_bf16(
          false, afrag, false, bfrag, (short)0, acc[s], false, false);
    }
  }

  float be = benc[f0 + ln];
#pragma unroll
  for (int s = 0; s < 8; ++s) {
    int bt = wave * 8 + s;
#pragma unroll
    for (int r = 0; r < 8; ++r) {
      int m = bt * 16 + lh * 8 + r;                  // C layout: row = vgpr + half*8
      pre[(size_t)m * NFEAT + f0 + ln] = acc[s][r] + be;
    }
  }
}

// ---------------------------------------------------------------- kernel 3
// One block per row: LDS-cached 32768-wide row, 32 argmax passes, then
// rewrite the coeffs row in place (zeros + relu'd winners).
__global__ void __launch_bounds__(256)
k_topk(float* __restrict__ coeffs, float* __restrict__ tkv, int* __restrict__ tki) {
  extern __shared__ float sm[];
  float* row  = sm;                          // 32768 f32 = 128 KB
  float* rmax = sm + NFEAT;                  // 256
  int*   rid  = (int*)(rmax + 256);          // 256
  float* wv   = (float*)(rid + 256);         // 32
  int*   wi   = (int*)(wv + 32);             // 32
  const int b = blockIdx.x, tid = threadIdx.x;
  float* prow = coeffs + (size_t)b * NFEAT;

  for (int e = tid; e < NFEAT; e += 256) row[e] = prow[e];
  __syncthreads();

  for (int it = 0; it < TOPK; ++it) {
    float bm = -3.4e38f; int bi = 0;
    const int base = tid * 128;
#pragma unroll 4
    for (int e = 0; e < 128; ++e) {
      float v = row[base + e];
      if (v > bm) { bm = v; bi = base + e; }
    }
    rmax[tid] = bm; rid[tid] = bi;
    __syncthreads();
    for (int s = 128; s > 0; s >>= 1) {
      if (tid < s && rmax[tid + s] > rmax[tid]) { rmax[tid] = rmax[tid + s]; rid[tid] = rid[tid + s]; }
      __syncthreads();
    }
    if (tid == 0) { wv[it] = rmax[0]; wi[it] = rid[0]; row[rid[0]] = -3.4e38f; }
    __syncthreads();
  }

  for (int e = tid; e < NFEAT; e += 256) prow[e] = 0.0f;
  __syncthreads();
  if (tid < TOPK) {
    float v = wv[tid] > 0.0f ? wv[tid] : 0.0f;       // relu on kept values
    prow[wi[tid]]        = v;
    tkv[b * TOPK + tid]  = v;
    tki[b * TOPK + tid]  = wi[tid];
  }
}

// ---------------------------------------------------------------- kernel 4
// One block per batch. recon[64x64] = (val*Vsel)^T[64x32] @ Wsel[32x64] + bias.
// K = 32 -> exactly one v_wmma_f32_16x16x32_f16 per 16x16 tile; MSE fused.
__global__ void __launch_bounds__(256)
k_decode(const float* __restrict__ x, const float* __restrict__ Vdec,
         const float* __restrict__ Wdec, const float* __restrict__ bias,
         const float* __restrict__ tkv, const int* __restrict__ tki,
         float* __restrict__ recon, float* __restrict__ mse) {
  __shared__ _Float16 As [64][32];   // As[m][j]  = val_j * V_dec[idx_j][m]   (K-minor)
  __shared__ _Float16 Bst[64][32];   // Bst[n][j] = W_dec[idx_j][n]           (K-minor)
  __shared__ float vals[TOPK];
  __shared__ int   idxs[TOPK];
  __shared__ float red[256];
  const int b = blockIdx.x, tid = threadIdx.x;

  if (tid < TOPK) { vals[tid] = tkv[b * TOPK + tid]; idxs[tid] = tki[b * TOPK + tid]; }
  __syncthreads();
  for (int e = tid; e < 64 * 32; e += 256) {
    int m = e >> 5, j = e & 31;
    As [m][j] = (_Float16)(vals[j] * Vdec[(size_t)idxs[j] * DK + m]);
    Bst[m][j] = (_Float16)(Wdec[(size_t)idxs[j] * DV + m]);
  }
  __syncthreads();

  const int wave = tid >> 5, lane = tid & 31, ln = lane & 15, lh = lane >> 4;
  float ss = 0.0f;

#pragma unroll
  for (int tt = 0; tt < 2; ++tt) {                   // 8 waves x 2 = 16 tiles
    int t = wave + tt * 8;
    int tm = t >> 2, tn = t & 3;

    // A frag: dwords {lh*8 + j, j=0..7} contiguous, 16B aligned.
    int m = tm * 16 + ln;
    const uint4* aro = (const uint4*)(&As[m][0]);
    v16h a = frag_from2h(aro[lh * 2], aro[lh * 2 + 1]);

    // B frag: dwords {lh*8 + j, j=0..7} contiguous -> two b128 loads.
    int n = tn * 16 + ln;
    const uint4* bro = (const uint4*)(&Bst[n][0]);
    v16h bb = frag_from2h(bro[lh * 2], bro[lh * 2 + 1]);

    v8f c = {};
    c = __builtin_amdgcn_wmma_f32_16x16x32_f16(false, a, false, bb, (short)0, c, false, false);

#pragma unroll
    for (int r = 0; r < 8; ++r) {
      int mm = tm * 16 + lh * 8 + r;
      float o = c[r] + bias[mm * 64 + n];
      size_t oidx = (size_t)b * DKV + mm * 64 + n;
      recon[oidx] = o;
      float d = o - x[oidx];
      ss += d * d;
    }
  }

  red[tid] = ss;
  __syncthreads();
  for (int s = 128; s > 0; s >>= 1) { if (tid < s) red[tid] += red[tid + s]; __syncthreads(); }
  if (tid == 0) atomicAdd(mse, red[0] * (1.0f / (float)(BATCH * DKV)));
}

// ---------------------------------------------------------------- launch
extern "C" void kernel_launch(void* const* d_in, const int* in_sizes, int n_in,
                              void* d_out, int out_size, void* d_ws, size_t ws_size,
                              hipStream_t stream) {
  const float* x    = (const float*)d_in[0];
  const float* Venc = (const float*)d_in[1];
  const float* Wenc = (const float*)d_in[2];
  const float* benc = (const float*)d_in[3];
  const float* Vdec = (const float*)d_in[4];
  const float* Wdec = (const float*)d_in[5];
  const float* bias = (const float*)d_in[6];

  float* out    = (float*)d_out;
  float* recon  = out;                                          // [1024*4096]
  float* coeffs = out + (size_t)BATCH * DKV;                    // [1024*32768]
  float* mse    = coeffs + (size_t)BATCH * NFEAT;               // [1]

  unsigned short* xbf = (unsigned short*)d_ws;                  // 8 MB
  float* tkv = (float*)((char*)d_ws + (size_t)BATCH * DKV * 2);
  int*   tki = (int*)((char*)tkv + (size_t)BATCH * TOPK * 4);

  k_convert_x<<<(BATCH * DKV + 255) / 256, 256, 0, stream>>>(x, xbf, mse);

  k_encode<<<NFEAT / 16, 256, 16 * DKV * sizeof(unsigned short), stream>>>(
      xbf, Venc, Wenc, benc, coeffs);

  size_t topk_lds = (size_t)(NFEAT + 256 + 256 + TOPK + TOPK) * 4;
  k_topk<<<BATCH, 256, topk_lds, stream>>>(coeffs, tkv, tki);

  k_decode<<<BATCH, 256, 0, stream>>>(x, Vdec, Wdec, bias, tkv, tki, recon, mse);
}
